// StereoCNN_47562467836252
// MI455X (gfx1250) — matrix operand order repaired and checked by reading, not compiled, not run on hardware
//
#include <hip/hip_runtime.h>

// ---------------------------------------------------------------------------
// StereoCNN for MI455X (gfx1250, wave32, WMMA).
// bf16 activations NHWC, f32 accumulation via v_wmma_f32_16x16x32_bf16.
// conv64: implicit GEMM, 32px x 64cout tile per wave (2 WMMAs per B fragment
//         -> halves LDS bandwidth per FLOP vs 16px tiles; LDS-staged weights).
// Correlation = banded Gram matrix via WMMA; softmax via LDS + shfl reductions.
// Workspace: 3 x 64MB activation buffers + repacked weights (~193 MB of d_ws).
// ---------------------------------------------------------------------------

typedef __bf16 bf16;
typedef __attribute__((ext_vector_type(16))) __bf16 v16bf;
typedef __attribute__((ext_vector_type(8)))  float  v8f;

#define BB 4
#define HH 256
#define WW 512
#define CC 64
#define DD 128

struct P32 { uint4 a, b; };  // 32 bytes == one v16bf fragment

// Load a v16bf fragment: 8 contiguous bf16 at p, 8 more at p+gap (elements).
// A-fragments use gap=16 (K layout split), B-fragments use gap=16 too (same K order).
__device__ __forceinline__ v16bf load_frag(const bf16* p, int gap) {
    P32 r;
    r.a = *reinterpret_cast<const uint4*>(p);
    r.b = *reinterpret_cast<const uint4*>(p + gap);
    return __builtin_bit_cast(v16bf, r);
}

__device__ __forceinline__ v16bf zero_frag() {
    P32 r;
    r.a = make_uint4(0u, 0u, 0u, 0u);
    r.b = make_uint4(0u, 0u, 0u, 0u);
    return __builtin_bit_cast(v16bf, r);
}

__device__ __forceinline__ v8f wmma_bf16(v16bf a, v16bf b, v8f c) {
    return __builtin_amdgcn_wmma_f32_16x16x32_bf16(false, a, false, b,
                                                   (short)0, c, false, false);
}

// ---------------------------------------------------------------------------
// Weight repack kernels
// ---------------------------------------------------------------------------

// [O=64,I=3,3,3] f32 (x2 towers) -> [tower][o][k]: k = c*9+ty*3+tx, padded to 32
__global__ void repack_w0(const float* __restrict__ lw0,
                          const float* __restrict__ rw0,
                          bf16* __restrict__ dst) {
    int idx = blockIdx.x * 256 + threadIdx.x;          // 2*64*32 = 4096
    int tower = idx >> 11, rem = idx & 2047;
    int o = rem >> 5, k = rem & 31;
    float v = 0.f;
    if (k < 27) {
        int c = k / 9, r9 = k % 9, ty = r9 / 3, tx = r9 % 3;
        const float* src = tower ? rw0 : lw0;
        v = src[((o * 3 + c) * 3 + ty) * 3 + tx];
    }
    dst[idx] = (bf16)v;
}

// [L=4,O=64,I=64,3,3] f32 (x2 towers) -> [tower*4+L][tap][o][i] bf16
__global__ void repack_w64(const float* __restrict__ lw,
                           const float* __restrict__ rw,
                           bf16* __restrict__ dst) {
    int idx = blockIdx.x * 256 + threadIdx.x;          // 2*4*9*64*64 = 294912
    int tower = idx / 147456, r = idx % 147456;
    int layer = r / 36864;  r %= 36864;
    int tap   = r / 4096;   r %= 4096;
    int o = r >> 6, i = r & 63;
    int ty = tap / 3, tx = tap % 3;
    const float* src = tower ? rw : lw;
    float v = src[(((layer * 64 + o) * 64 + i) * 3 + ty) * 3 + tx];
    dst[((tower * 4 + layer) * 9 + tap) * 4096 + o * 64 + i] = (bf16)v;
}

// ---------------------------------------------------------------------------
// Layer 0: 3 -> 64 channels, SAME 3x3, ReLU. Single K=32 WMMA (27 taps padded).
// Block: 256 threads = 8 waves; block covers (b,h) and 128 consecutive w.
// ---------------------------------------------------------------------------
__global__ void conv0_wmma(const float* __restrict__ in,   // [B,3,H,W] f32
                           const bf16*  __restrict__ wp,   // [64][32] bf16
                           const float* __restrict__ bias, // [64]
                           bf16* __restrict__ out) {       // NHWC bf16
    __shared__ alignas(16) float sPatch[3 * 3 * 132];      // [c][row][x -1..128]
    __shared__ alignas(16) bf16  sW[64 * 32];

    const int tid  = threadIdx.x;
    const int wgrp = blockIdx.x & 3;
    const int h    = (blockIdx.x >> 2) & 255;
    const int b    = blockIdx.x >> 10;
    const int wblk = wgrp * 128;

    // Stage weights (4096 B: one uint4 per thread)
    reinterpret_cast<uint4*>(sW)[tid] = reinterpret_cast<const uint4*>(wp)[tid];

    // Stage zero-padded input patch: 3 ch x 3 rows x 130 w
    for (int i = tid; i < 3 * 3 * 130; i += 256) {
        int c = i / 390, rr = (i / 130) % 3, xx = i % 130;
        int hh = h - 1 + rr, ww2 = wblk - 1 + xx;
        float v = 0.f;
        if (hh >= 0 && hh < HH && ww2 >= 0 && ww2 < WW)
            v = in[((size_t)(b * 3 + c) * HH + hh) * WW + ww2];
        sPatch[(c * 3 + rr) * 132 + xx] = v;
    }
    __syncthreads();

    const int lane = tid & 31, wv = tid >> 5;
    const int m = lane & 15, hi = lane >> 4;
    const int w0 = wblk + wv * 16;

    // A fragment: 16 pixels x K=32 (K = c*9 + ty*3 + tx, zeros for K>=27)
    v16bf a;
#pragma unroll
    for (int e = 0; e < 16; ++e) {
        int k = hi * 8 + e + ((e & 8) ? 8 : 0);
        float f = 0.f;
        if (k < 27) {
            int c = k / 9, r9 = k % 9, rr = r9 / 3, tx = r9 % 3;
            f = sPatch[(c * 3 + rr) * 132 + (wv * 16 + m + tx)];
        }
        a[e] = (bf16)f;
    }

    const int koff = hi * 8;
    bf16* op = out + ((size_t)(b * HH + h) * WW) * CC;
#pragma unroll
    for (int j = 0; j < 4; ++j) {
        v16bf bb = load_frag(&sW[(j * 16 + m) * 32 + koff], 16);
        v8f acc = {0.f, 0.f, 0.f, 0.f, 0.f, 0.f, 0.f, 0.f};
        acc = wmma_bf16(a, bb, acc);
        float bi = bias[j * 16 + m];
#pragma unroll
        for (int r = 0; r < 8; ++r) {
            int mm = r + hi * 8;
            float v = fmaxf(acc[r] + bi, 0.f);
            op[(size_t)(w0 + mm) * CC + j * 16 + m] = (bf16)v;
        }
    }
}

// ---------------------------------------------------------------------------
// 64 -> 64 conv, SAME 3x3, optional ReLU. Implicit GEMM.
// Wave tile: 32(w) x 64(cout); K = 9 taps x 2x32 cin. Each LDS B fragment
// feeds 2 WMMAs (two M sub-tiles) -> LDS bytes/FLOP halved vs 16px tiles.
// Block: 256 threads = 8 waves, covers (b,h) and 256 consecutive w.
// ---------------------------------------------------------------------------
__global__ void conv64_wmma(const bf16* __restrict__ in,    // NHWC bf16
                            const bf16* __restrict__ wgt,   // [tap][o][i] bf16
                            const float* __restrict__ bias, // [64]
                            bf16* __restrict__ out,         // NHWC bf16
                            int relu) {
    __shared__ alignas(16) bf16 sW[9 * 64 * 64];            // 72 KB

    const int tid  = threadIdx.x;
    const int wgrp = blockIdx.x & 1;
    const int h    = (blockIdx.x >> 1) & 255;
    const int b    = blockIdx.x >> 9;

    // Stage all layer weights: 4608 uint4, 18 per thread
    {
        uint4* d = reinterpret_cast<uint4*>(sW);
        const uint4* s = reinterpret_cast<const uint4*>(wgt);
        for (int i = tid; i < 4608; i += 256) d[i] = s[i];
    }
    __syncthreads();

    const int lane = tid & 31, wv = tid >> 5;
    const int m = lane & 15, hi = lane >> 4;
    const int w0 = wgrp * 256 + wv * 32;                // 32-pixel M tile
    const size_t rowbase = (size_t)(b * HH) * WW * CC;

    v8f acc[2][4];
#pragma unroll
    for (int u = 0; u < 2; ++u)
#pragma unroll
        for (int j = 0; j < 4; ++j)
            acc[u][j] = {0.f,0.f,0.f,0.f,0.f,0.f,0.f,0.f};

    for (int tap = 0; tap < 9; ++tap) {
        int dy = tap / 3 - 1, dx = tap % 3 - 1;
        int hh = h + dy;
        if (hh < 0 || hh >= HH) continue;               // uniform
        const size_t rb = rowbase + (size_t)hh * WW * CC;
        int  ww2[2];  bool ok[2];
        const bf16* ap[2];
#pragma unroll
        for (int u = 0; u < 2; ++u) {
            ww2[u] = w0 + u * 16 + m + dx;               // per-lane
            ok[u]  = (ww2[u] >= 0) && (ww2[u] < WW);
            ap[u]  = in + rb + (size_t)(ok[u] ? ww2[u] : 0) * CC;
        }
#pragma unroll
        for (int kc = 0; kc < 2; ++kc) {
            v16bf a0 = ok[0] ? load_frag(ap[0] + kc * 32 + hi * 8, 16) : zero_frag();
            v16bf a1 = ok[1] ? load_frag(ap[1] + kc * 32 + hi * 8, 16) : zero_frag();
#pragma unroll
            for (int j = 0; j < 4; ++j) {
                v16bf bb = load_frag(
                    &sW[(tap * 64 + j * 16 + m) * 64 + kc * 32 + hi * 8], 16);
                acc[0][j] = wmma_bf16(a0, bb, acc[0][j]);
                acc[1][j] = wmma_bf16(a1, bb, acc[1][j]);
            }
        }
    }

    bf16* op = out + ((size_t)(b * HH + h) * WW) * CC;
#pragma unroll
    for (int u = 0; u < 2; ++u) {
#pragma unroll
        for (int j = 0; j < 4; ++j) {
            float bi = bias[j * 16 + m];
#pragma unroll
            for (int r = 0; r < 8; ++r) {
                int mm = r + hi * 8;
                float v = acc[u][j][r] + bi;
                if (relu) v = fmaxf(v, 0.f);
                op[(size_t)(w0 + u * 16 + mm) * CC + j * 16 + m] = (bf16)v;
            }
        }
    }
}

// ---------------------------------------------------------------------------
// Correlation band (WMMA Gram tiles) + softmax over disparity.
// Block: 128 threads = 4 waves; wave handles one 16-wide w tile of one (b,h).
// corr tile t: C[m][n] = sum_c L[c, w0+m] * R[c, w0-16t+n]; d = 16t + m - n.
// ---------------------------------------------------------------------------
__global__ void corr_softmax(const bf16* __restrict__ phiL,
                             const bf16* __restrict__ phiR,
                             float* __restrict__ out) {
    __shared__ float sC[4][16][132];                    // [wave][w][d], 33.8 KB

    const int tid  = threadIdx.x;
    const int lane = tid & 31, wv = tid >> 5;
    const int m = lane & 15, hi = lane >> 4;
    const int wgrp = blockIdx.x & 7;
    const int h    = (blockIdx.x >> 3) & 255;
    const int b    = blockIdx.x >> 11;
    const int w0   = wgrp * 64 + wv * 16;

    const size_t rowbase = (size_t)(b * HH + h) * WW * CC;
    const bf16* ap = phiL + rowbase + (size_t)(w0 + m) * CC;
    v16bf a0 = load_frag(ap + hi * 8, 16);              // K = 0..31
    v16bf a1 = load_frag(ap + 32 + hi * 8, 16);         // K = 32..63

    for (int t = 0; t < 9; ++t) {
        int wp = w0 - 16 * t + m;                       // B column (lane = n)
        bool ok = (wp >= 0);
        const bf16* bp = phiR + rowbase + (size_t)(ok ? wp : 0) * CC;
        v16bf b0 = ok ? load_frag(bp + hi * 8, 16)      : zero_frag();
        v16bf b1 = ok ? load_frag(bp + 32 + hi * 8, 16) : zero_frag();
        v8f acc = {0.f,0.f,0.f,0.f,0.f,0.f,0.f,0.f};
        acc = wmma_bf16(a0, b0, acc);
        acc = wmma_bf16(a1, b1, acc);
#pragma unroll
        for (int r = 0; r < 8; ++r) {
            int mm = r + hi * 8;
            int d = 16 * t + mm - m;                    // n == m (lane&15)
            if (d >= 0 && d < DD) sC[wv][mm][d] = acc[r];
        }
    }
    __syncthreads();

    // Softmax over d per (w). lane&15 = row, hi splits the 128 cols in half.
    const int cbase = hi * 64;
    float mx = -3.0e38f;
    for (int c = 0; c < 64; ++c) mx = fmaxf(mx, sC[wv][m][cbase + c]);
    mx = fmaxf(mx, __shfl_xor(mx, 16, 32));
    float s = 0.f;
    for (int c = 0; c < 64; ++c) {
        float e = __expf(sC[wv][m][cbase + c] - mx);
        sC[wv][m][cbase + c] = e;
        s += e;
    }
    s += __shfl_xor(s, 16, 32);
    float inv = 1.f / s;
    __syncthreads();

    // Coalesced write: lanes 0-15 -> plane d, lanes 16-31 -> plane d+1.
    float* ob = out + ((size_t)b * DD * HH + h) * WW + w0 + m;
    for (int d = 0; d < DD; d += 2) {
        int dd = d + hi;
        ob[(size_t)dd * HH * WW] = sC[wv][m][dd] * inv;
    }
}

// ---------------------------------------------------------------------------
extern "C" void kernel_launch(void* const* d_in, const int* in_sizes, int n_in,
                              void* d_out, int out_size, void* d_ws, size_t ws_size,
                              hipStream_t stream) {
    const float* l   = (const float*)d_in[0];
    const float* r   = (const float*)d_in[1];
    const float* lw0 = (const float*)d_in[2];
    const float* lb0 = (const float*)d_in[3];
    const float* lw  = (const float*)d_in[4];
    const float* lb  = (const float*)d_in[5];
    const float* rw0 = (const float*)d_in[6];
    const float* rb0 = (const float*)d_in[7];
    const float* rw  = (const float*)d_in[8];
    const float* rb  = (const float*)d_in[9];
    float* outp = (float*)d_out;

    const size_t ACT = (size_t)BB * HH * WW * CC;       // 33.5M bf16 = 64 MB
    bf16* bufA = (bf16*)d_ws;
    bf16* bufB = bufA + ACT;
    bf16* bufC = bufB + ACT;
    bf16* wW64 = bufC + ACT;                            // 294912 bf16
    bf16* wW0  = wW64 + (size_t)2 * 4 * 9 * 64 * 64;    // 4096 bf16

    repack_w0 <<<16,   256, 0, stream>>>(lw0, rw0, wW0);
    repack_w64<<<1152, 256, 0, stream>>>(lw, rw, wW64);

    // Left tower: l -> A -> B -> A -> B -> C (phiL = C)
    conv0_wmma <<<4096, 256, 0, stream>>>(l, wW0, lb0, bufA);
    conv64_wmma<<<2048, 256, 0, stream>>>(bufA, wW64 + 0 * 36864, lb +   0, bufB, 1);
    conv64_wmma<<<2048, 256, 0, stream>>>(bufB, wW64 + 1 * 36864, lb +  64, bufA, 1);
    conv64_wmma<<<2048, 256, 0, stream>>>(bufA, wW64 + 2 * 36864, lb + 128, bufB, 1);
    conv64_wmma<<<2048, 256, 0, stream>>>(bufB, wW64 + 3 * 36864, lb + 192, bufC, 0);

    // Right tower: r -> A -> B -> A -> B -> A (phiR = A)
    conv0_wmma <<<4096, 256, 0, stream>>>(r, wW0 + 2048, rb0, bufA);
    conv64_wmma<<<2048, 256, 0, stream>>>(bufA, wW64 + 4 * 36864, rb +   0, bufB, 1);
    conv64_wmma<<<2048, 256, 0, stream>>>(bufB, wW64 + 5 * 36864, rb +  64, bufA, 1);
    conv64_wmma<<<2048, 256, 0, stream>>>(bufA, wW64 + 6 * 36864, rb + 128, bufB, 1);
    conv64_wmma<<<2048, 256, 0, stream>>>(bufB, wW64 + 7 * 36864, rb + 192, bufA, 0);

    corr_softmax<<<8192, 128, 0, stream>>>(bufC, bufA, outp);
}
